// SelfAttention_14826227105871
// MI455X (gfx1250) — matrix operand order repaired and checked by reading, not compiled, not run on hardware
//
#include <hip/hip_runtime.h>
#include <hip/hip_bf16.h>

// MI455X / gfx1250 fused self-attention:
//   prep:  g = sqrt(log2 e) * (W*x + b)  (bf16, [B][N][32]),  x -> bf16 [B][C][N]
//   attn:  per-WG double-buffered async staging (global_load_async_to_lds_b128)
//          of keys/values into LDS; scores = g^T g (WMMA bf16, base-2 domain);
//          per-lane online softmax, one final cross-lane reduction;
//          alpha -> HBM (512 MB = roofline floor); r = alpha * x^T (WMMA bf16);
//          residual + ReLU. 32 queries per wave, 8 waves/WG sharing the stream.

typedef __attribute__((ext_vector_type(16))) __bf16 v16bf;
typedef __attribute__((ext_vector_type(8)))  float  v8f;

#define B_    8
#define C_IN  64
#define C_A   32
#define N_    4096
#define CHUNK 128   // keys staged per LDS buffer

union FragB { v16bf v; unsigned int u[8]; };
union Acc   { v8f   v; float        f[8]; };

__device__ inline unsigned short f32_to_bf16_bits(float f) {
    union { float f; unsigned int u; } x; x.f = f;
    unsigned int r = x.u + 0x7FFFu + ((x.u >> 16) & 1u);   // RNE
    return (unsigned short)(r >> 16);
}

__device__ inline unsigned lds_addr(const void* p) {
    return (unsigned)(size_t)p;            // generic LDS ptr: low 32 bits = LDS offset
}
__device__ inline void async_copy_b128(unsigned lds_dst, const void* gsrc) {
    asm volatile("global_load_async_to_lds_b128 %0, %1, off"
                 :: "v"(lds_dst), "v"((unsigned long long)(size_t)gsrc) : "memory");
}

// ---------------------------------------------------------------------------
// Kernel 1: 1x1 conv (g = W x + b), pre-scaled by sqrt(log2 e) so the score
// GEMM lands directly in base-2 domain; also x f32 -> bf16.
// ---------------------------------------------------------------------------
__global__ __launch_bounds__(256) void prep_kernel(
    const float* __restrict__ x, const float* __restrict__ W,
    const float* __restrict__ bias,
    unsigned short* __restrict__ g16,   // [B][N][C_A] bf16 (scaled)
    unsigned short* __restrict__ x16)   // [B][C_IN][N] bf16
{
    __shared__ float sW[C_A * C_IN];
    __shared__ float sB[C_A];
    for (int i = threadIdx.x; i < C_A * C_IN; i += blockDim.x) sW[i] = W[i];
    if (threadIdx.x < C_A) sB[threadIdx.x] = bias[threadIdx.x];
    __syncthreads();

    int id = blockIdx.x * blockDim.x + threadIdx.x;   // 0 .. B*N-1
    int b = id >> 12;
    int n = id & (N_ - 1);

    float acc[C_A];
    #pragma unroll
    for (int c = 0; c < C_A; ++c) acc[c] = sB[c];

    #pragma unroll 4
    for (int ci = 0; ci < C_IN; ++ci) {
        float xv = x[((size_t)(b * C_IN + ci) << 12) + n];
        x16[((size_t)(b * C_IN + ci) << 12) + n] = f32_to_bf16_bits(xv);
        #pragma unroll
        for (int c = 0; c < C_A; ++c) acc[c] += sW[c * C_IN + ci] * xv;
    }

    const float GS = 1.2011224087864498f;   // sqrt(log2 e)
    unsigned short* grow = g16 + (size_t)(b * N_ + n) * C_A;
    #pragma unroll
    for (int c = 0; c < C_A; ++c) grow[c] = f32_to_bf16_bits(acc[c] * GS);
}

// ---------------------------------------------------------------------------
// Kernel 2: fused attention. One wave per 32-query tile; 8 waves per WG share
// the key/value stream through double-buffered async LDS staging.
// ---------------------------------------------------------------------------
__global__ __launch_bounds__(256, 1) void attn_kernel(
    const unsigned short* __restrict__ g16,   // [B][N][32] bf16 (scaled)
    const unsigned short* __restrict__ x16,   // [B][64][N] bf16
    const float* __restrict__ x,              // [B][64][N] f32 (residual)
    float* __restrict__ out,                  // [B][64][N]
    float* __restrict__ alpha)                // [B][N][N]
{
    __shared__ unsigned short sG[2][CHUNK * C_A];      // 2 x 8 KB keys
    __shared__ unsigned short sX[2][C_IN * CHUNK];     // 2 x 16 KB values
    __shared__ unsigned short sAlpha[8][2][16 * 32];   // 16 KB D->A staging

    const int tid  = threadIdx.x;
    const int lane = tid & 31;
    const int wave = tid >> 5;
    const int tile = blockIdx.x * 8 + wave;          // 0..1023
    const int b    = tile >> 7;
    const int n0   = (tile & 127) << 5;              // 32 queries
    const int col  = lane & 15;
    const int half = lane >> 4;

    const unsigned short* gSrc = g16 + (size_t)b * N_ * C_A;
    const unsigned short* xSrc = x16 + (size_t)b * C_IN * N_;

    // --- cooperative async copies (whole WG) ---------------------------------
    // g chunk: 8 KB linear; 256 threads x 32 B (2 x b128)
    auto issueG = [&](int buf, int m0) {
        const unsigned short* src = gSrc + (size_t)m0 * C_A + tid * 16;
        unsigned dst = lds_addr(&sG[buf][tid * 16]);
        async_copy_b128(dst,       src);
        async_copy_b128(dst + 16u, src + 8);
    };
    // x chunk: 64 rows x 256 B; thread -> (row c = tid/4, quarter q = tid%4), 4 x b128
    auto issueX = [&](int buf, int m0) {
        int c = tid >> 2, q = tid & 3;
        const unsigned short* src = xSrc + (size_t)c * N_ + m0 + q * 32;
        unsigned dst = lds_addr(&sX[buf][c * CHUNK + q * 32]);
        #pragma unroll
        for (int i = 0; i < 4; ++i)
            async_copy_b128(dst + 16u * i, src + 8 * i);
    };

    // --- fragment loaders ----------------------------------------------------
    // A 16x32 bf16: lane row M = lane%16; VGPR v K-pair K = 2*v+16*half
    auto loadAfrag = [&](FragB& f, const unsigned int* row32) {
        #pragma unroll
        for (int v = 0; v < 8; ++v) {
            int k = 2 * v + 16 * half;
            f.u[v] = row32[k >> 1];
        }
    };
    // B 32x16 bf16 from LDS key buffer: lane col = lane%16; K-pair K = 2v+16*half
    auto loadBfragG = [&](FragB& f, const unsigned short* base, int mloc) {
        const unsigned int* row = (const unsigned int*)(base + (mloc + col) * C_A) + 8 * half;
        #pragma unroll
        for (int v = 0; v < 8; ++v) f.u[v] = row[v];
    };
    // B 32x16 bf16 from LDS value buffer (row-major [c][CHUNK])
    auto loadBfragX = [&](FragB& f, const unsigned short* base, int c, int mloc) {
        const unsigned int* row = (const unsigned int*)(base + c * CHUNK + mloc) + 8 * half;
        #pragma unroll
        for (int v = 0; v < 8; ++v) f.u[v] = row[v];
    };

    // Two query A-fragments (rows n0..n0+15, n0+16..n0+31), persistent in VGPRs.
    FragB aG[2];
    #pragma unroll
    for (int qh = 0; qh < 2; ++qh)
        loadAfrag(aG[qh], (const unsigned int*)(gSrc + (size_t)(n0 + qh * 16 + col) * C_A));

    // ---------------- Pass 1: per-lane online row max / sum ------------------
    float runMax[2][8], runSum[2][8];
    #pragma unroll
    for (int qh = 0; qh < 2; ++qh)
        #pragma unroll
        for (int v = 0; v < 8; ++v) { runMax[qh][v] = -1.0e38f; runSum[qh][v] = 0.0f; }

    issueG(0, 0);
    asm volatile("s_wait_asynccnt 0" ::: "memory");
    __syncthreads();

    int cur = 0;
    for (int m0 = 0; m0 < N_; m0 += CHUNK) {
        if (m0 + CHUNK < N_) issueG(cur ^ 1, m0 + CHUNK);   // overlap copy w/ compute

        #pragma unroll
        for (int grp = 0; grp < 2; ++grp) {                 // 2 x 64 keys
            FragB bG[4];
            #pragma unroll
            for (int t = 0; t < 4; ++t) loadBfragG(bG[t], sG[cur], grp * 64 + t * 16);
            #pragma unroll
            for (int qh = 0; qh < 2; ++qh) {
                Acc s[4];
                #pragma unroll
                for (int t = 0; t < 4; ++t) {
                    v8f z = {};
                    s[t].v = __builtin_amdgcn_wmma_f32_16x16x32_bf16(
                        false, aG[qh].v, false, bG[t].v, (short)0, z, false, false);
                }
                #pragma unroll
                for (int v = 0; v < 8; ++v) {
                    float mx = fmaxf(fmaxf(s[0].f[v], s[1].f[v]),
                                     fmaxf(s[2].f[v], s[3].f[v]));
                    float nm = fmaxf(runMax[qh][v], mx);
                    float e0 = exp2f(s[0].f[v] - nm);
                    float e1 = exp2f(s[1].f[v] - nm);
                    float e2 = exp2f(s[2].f[v] - nm);
                    float e3 = exp2f(s[3].f[v] - nm);
                    runSum[qh][v] = runSum[qh][v] * exp2f(runMax[qh][v] - nm)
                                  + ((e0 + e1) + (e2 + e3));
                    runMax[qh][v] = nm;
                }
            }
        }
        asm volatile("s_wait_asynccnt 0" ::: "memory");
        __syncthreads();
        cur ^= 1;
    }

    // Single cross-lane (16-lane half) reduction: row max, rescaled row sum.
    float invSum[2][8];
    #pragma unroll
    for (int qh = 0; qh < 2; ++qh)
        #pragma unroll
        for (int v = 0; v < 8; ++v) {
            float gm = runMax[qh][v];
            #pragma unroll
            for (int off = 1; off < 16; off <<= 1)
                gm = fmaxf(gm, __shfl_xor(gm, off, 32));
            float sc = runSum[qh][v] * exp2f(runMax[qh][v] - gm);
            #pragma unroll
            for (int off = 1; off < 16; off <<= 1)
                sc += __shfl_xor(sc, off, 32);
            runMax[qh][v] = gm;
            invSum[qh][v] = 1.0f / sc;
        }

    // ---------------- Pass 2: alpha out + r = alpha * x^T --------------------
    Acc r[2][4];
    #pragma unroll
    for (int qh = 0; qh < 2; ++qh)
        #pragma unroll
        for (int t = 0; t < 4; ++t) { v8f z = {}; r[qh][t].v = z; }

    float* alphaB = alpha + (size_t)b * N_ * N_;

    issueG(0, 0);
    issueX(0, 0);
    asm volatile("s_wait_asynccnt 0" ::: "memory");
    __syncthreads();

    cur = 0;
    for (int m0 = 0; m0 < N_; m0 += CHUNK) {
        if (m0 + CHUNK < N_) { issueG(cur ^ 1, m0 + CHUNK); issueX(cur ^ 1, m0 + CHUNK); }

        #pragma unroll
        for (int sb = 0; sb < 4; ++sb) {                    // 4 x 32-key sub-blocks
            int ml = sb * 32;
            FragB bG0, bG1;
            loadBfragG(bG0, sG[cur], ml);
            loadBfragG(bG1, sG[cur], ml + 16);

            #pragma unroll
            for (int qh = 0; qh < 2; ++qh) {
                unsigned short* aLds = sAlpha[wave][qh];
                #pragma unroll
                for (int t = 0; t < 2; ++t) {
                    v8f z = {};
                    Acc s;
                    s.v = __builtin_amdgcn_wmma_f32_16x16x32_bf16(
                        false, aG[qh].v, false, (t ? bG1.v : bG0.v), (short)0, z,
                        false, false);
                    #pragma unroll
                    for (int v = 0; v < 8; ++v) {
                        int row = v + 8 * half;
                        float a = exp2f(s.f[v] - runMax[qh][v]) * invSum[qh][v];
                        alphaB[(size_t)(n0 + qh * 16 + row) * N_ +
                               (m0 + ml + t * 16 + col)] = a;
                        aLds[row * 32 + t * 16 + col] = f32_to_bf16_bits(a);
                    }
                }
            }
            // wave-local LDS RAW: alpha stores feed the A-fragment loads below
            asm volatile("s_wait_dscnt 0" ::: "memory");

            FragB aA[2];
            #pragma unroll
            for (int qh = 0; qh < 2; ++qh)
                loadAfrag(aA[qh], (const unsigned int*)sAlpha[wave][qh] + col * 16);

            #pragma unroll
            for (int t = 0; t < 4; ++t) {        // c-tiles; B-frag shared by both qh
                FragB bX;
                loadBfragX(bX, sX[cur], t * 16 + col, ml);
                #pragma unroll
                for (int qh = 0; qh < 2; ++qh)
                    r[qh][t].v = __builtin_amdgcn_wmma_f32_16x16x32_bf16(
                        false, aA[qh].v, false, bX.v, (short)0, r[qh][t].v,
                        false, false);
            }
        }
        asm volatile("s_wait_asynccnt 0" ::: "memory");
        __syncthreads();
        cur ^= 1;
    }

    // ---------------- Epilogue: residual + ReLU ------------------------------
    const float* xfB  = x   + (size_t)b * C_IN * N_;
    float*       outB = out + (size_t)b * C_IN * N_;
    #pragma unroll
    for (int qh = 0; qh < 2; ++qh)
        #pragma unroll
        for (int t = 0; t < 4; ++t) {
            int c = t * 16 + col;
            #pragma unroll
            for (int v = 0; v < 8; ++v) {
                int n = n0 + qh * 16 + v + 8 * half;
                float val = r[qh][t].f[v] + xfB[(size_t)c * N_ + n];
                outB[(size_t)c * N_ + n] = fmaxf(val, 0.0f);
            }
        }
}

// ---------------------------------------------------------------------------
extern "C" void kernel_launch(void* const* d_in, const int* in_sizes, int n_in,
                              void* d_out, int out_size, void* d_ws, size_t ws_size,
                              hipStream_t stream) {
    const float* x    = (const float*)d_in[0];   // [8, 64, 64, 64]
    const float* W    = (const float*)d_in[1];   // [32, 64]
    const float* bias = (const float*)d_in[2];   // [32]

    float* out   = (float*)d_out;                       // [8, 64, 64, 64]
    float* alpha = out + (size_t)B_ * C_IN * N_;        // [8, 4096, 4096]

    unsigned short* g16 = (unsigned short*)d_ws;                    // 2 MB
    unsigned short* x16 = g16 + (size_t)B_ * N_ * C_A;              // 4 MB

    prep_kernel<<<(B_ * N_) / 256, 256, 0, stream>>>(x, W, bias, g16, x16);
    attn_kernel<<<(B_ * (N_ / 32)) / 8, 256, 0, stream>>>(g16, x16, x, out, alpha);
}
